// MultiDiscretePolicy_64338610094525
// MI455X (gfx1250) — matrix operand order repaired and checked by reading, not compiled, not run on hardware
//
#include <hip/hip_runtime.h>
#include <stdint.h>

// ---------------------------------------------------------------------------
// MultiDiscretePolicy forward for MI455X (gfx1250, wave32, WMMA).
//   GEMM1: h  = relu(s @ W1 + b1)        [32768,1024] x [1024,1024]
//   GEMM2: aw = h @ W2 + b2              [32768,1024] x [1024,256]
//   GEMM3: z  = aw @ W3 + head_b         [32768,256]  x [256,1024]
//   out[b,2h+k] = one_hot(argmax_k(z + gumbel(u)))   (straight-through fwd)
// bf16 WMMA (v_wmma_f32_16x16x32_bf16), fp32 accum.
// Double-buffered async global->LDS staging (ASYNCcnt partial waits).
// Weights pre-packed in WMMA B-fragment order so every LDS access is b128.
// ---------------------------------------------------------------------------

typedef __attribute__((ext_vector_type(16))) __bf16 v16bf;
typedef __attribute__((ext_vector_type(8)))  float  v8f;

union FragBF { v16bf v; uint32_t u[8]; uint4 q[2]; };
union FragF  { v8f   v; float    f[8]; };

__device__ __forceinline__ uint32_t pack_bf16x2(float a, float b) {
  uint32_t ua = __float_as_uint(a), ub = __float_as_uint(b);
  uint32_t ra = (ua + 0x7FFFu + ((ua >> 16) & 1u)) >> 16;  // RNE
  uint32_t rb = (ub + 0x7FFFu + ((ub >> 16) & 1u)) >> 16;
  return (rb << 16) | (ra & 0xFFFFu);
}

__device__ __forceinline__ uint16_t f32_to_bf16(float a) {
  uint32_t ua = __float_as_uint(a);
  return (uint16_t)((ua + 0x7FFFu + ((ua >> 16) & 1u)) >> 16);
}

// Async copy 16B global -> LDS (per lane). ASYNCcnt-tracked; no VGPR data.
__device__ __forceinline__ void async_load_b128(uint32_t loff, const void* base,
                                                uint32_t goff) {
  asm volatile("global_load_async_to_lds_b128 %0, %1, %2"
               :: "v"(loff), "v"(goff), "s"(base) : "memory");
}
// ASYNCcnt decrements in issue order per wave -> partial waits give ping-pong.
__device__ __forceinline__ void wait_async_le0() { asm volatile("s_wait_asynccnt 0x0" ::: "memory"); }
__device__ __forceinline__ void wait_async_le2() { asm volatile("s_wait_asynccnt 0x2" ::: "memory"); }
__device__ __forceinline__ void wait_async_le4() { asm volatile("s_wait_asynccnt 0x4" ::: "memory"); }

// A fragment (16x32 bf16), contiguous K layout: lanes 0-15 (M=lane) K={0..15},
// lanes 16-31 (M=lane-16) K={16..31}; two K per VGPR.
__device__ __forceinline__ v16bf load_a_frag(const uint16_t* As16, int lda, int lane) {
  int m = lane & 15, half = lane >> 4;
  const uint16_t* p = As16 + m * lda + half * 16;
  FragBF f;
  f.q[0] = *reinterpret_cast<const uint4*>(p);
  f.q[1] = *reinterpret_cast<const uint4*>(p + 8);
  return f.v;
}

// B fragment from fragment-ordered LDS: block mc = [lane][12] uint slots
// (8 data words = pair-rows, 4 pad). Two contiguous b128 loads per lane.
__device__ __forceinline__ v16bf load_b_frag(const uint32_t* Bs, int mc, int lane) {
  const uint32_t* p = Bs + mc * 384 + lane * 12;
  FragBF f;
  f.q[0] = *reinterpret_cast<const uint4*>(p);
  f.q[1] = *reinterpret_cast<const uint4*>(p + 4);
  return f.v;
}

// ---------------------------------------------------------------------------
// Weight packing into WMMA B-fragment order:
//   out[((ks*(N/16) + nb)*32 + lane)*8 + word] = {bf16(W[2kp+1][n]), bf16(W[2kp][n])}
//   kp = k/2; ks = kp>>4; r = kp&15; lane = (n&15) + 16*(r>>3); word = r&7.
// ---------------------------------------------------------------------------
__device__ __forceinline__ size_t frag_index(int kp, int n, int N) {
  int ks = kp >> 4, r = kp & 15;
  int nb = n >> 4, nl = n & 15;
  int lane = nl + ((r >> 3) << 4);
  int word = r & 7;
  return (((size_t)(ks * (N >> 4) + nb) * 32 + lane) << 3) + word;
}

__global__ __launch_bounds__(256) void pack_w_kernel(const float* __restrict__ W,
                                                     uint32_t* __restrict__ Wf,
                                                     int K, int N) {
  int idx = blockIdx.x * 256 + threadIdx.x;
  if (idx >= (K / 2) * N) return;
  int kp = idx / N, n = idx - kp * N;
  float a = W[(size_t)(2 * kp) * N + n];
  float b = W[(size_t)(2 * kp + 1) * N + n];
  Wf[frag_index(kp, n, N)] = pack_bf16x2(a, b);
}

// head_W [512][256][2] -> W3 [256][1024]: W3[d][n] = head_W[n>>1][d][n&1]
__global__ __launch_bounds__(256) void pack_headw_kernel(const float* __restrict__ HW,
                                                         uint32_t* __restrict__ Wf) {
  int idx = blockIdx.x * 256 + threadIdx.x;
  if (idx >= 128 * 1024) return;
  int kp = idx >> 10, n = idx & 1023;
  int h = n >> 1, c = n & 1;
  float a = HW[(size_t)h * 512 + 4 * kp + c];        // d = 2kp
  float b = HW[(size_t)h * 512 + 4 * kp + 2 + c];    // d = 2kp+1
  Wf[frag_index(kp, n, 1024)] = pack_bf16x2(a, b);
}

// ---------------------------------------------------------------------------
// Tiled WMMA GEMM. MODE 0: fp32 A (convert), +bias, ReLU, bf16 out
//                  MODE 1: bf16 A (async),  +bias, bf16 out
//                  MODE 2: bf16 A (async),  +bias, gumbel-argmax out
// 256 threads = 8 waves, tile 128x128, K-step 32, wave owns 32x64.
// Double-buffered LDS; next tile's async loads stay in flight during compute.
// ---------------------------------------------------------------------------
template <int MODE>
__global__ __launch_bounds__(256) void gemm_wmma(const void* __restrict__ Aglob,
                                                 const uint32_t* __restrict__ Bp,
                                                 const float* __restrict__ bias,
                                                 void* __restrict__ Cout,
                                                 const float* __restrict__ Ug,
                                                 int K, int N) {
  constexpr int BM = 128, BN = 128, KT = 32;
  constexpr int LDA = KT + 8;    // bf16 units, padded
  constexpr int LDZ = BN + 2;    // fp32 staging (MODE 2)
  constexpr int LDO = BN + 8;    // bf16 staging (MODE 0/1)

  __shared__ uint16_t As[2][BM * LDA];
  __shared__ uint32_t Bs[2][8 * 384];  // 8 n-blocks x 32 lanes x 12-uint slots
  __shared__ float    zbuf[(MODE == 2) ? (BM * LDZ) : 1];
  __shared__ uint16_t obuf[(MODE <= 1) ? (BM * LDO) : 1];

  const int bm = blockIdx.x * BM;
  const int bn = blockIdx.y * BN;
  const int t = threadIdx.x;
  const int lane = t & 31, wave = t >> 5;
  const int wm = (wave & 3) * 32;
  const int wn = (wave >> 2) * 64;

  v8f acc[2][4];
#pragma unroll
  for (int i = 0; i < 2; ++i)
#pragma unroll
    for (int j = 0; j < 4; ++j)
      acc[i][j] = v8f{0.f, 0.f, 0.f, 0.f, 0.f, 0.f, 0.f, 0.f};

  // stage tile `ks` of A and B into LDS buffer `buf`
  auto stage = [&](int ks, int buf) {
    const int k0 = ks * KT;
    // B tile: 8 fragment blocks, contiguous async b128 copies (2 per thread)
#pragma unroll
    for (int p = 0; p < 2; ++p) {
      int idx4 = t + p * 256;          // 512 x uint4 = 8KB tile
      int mc = idx4 >> 6;              // n-block 0..7
      int rem = idx4 & 63;
      int ln = rem >> 1;               // lane slot 0..31
      int w4 = (rem & 1) * 4;          // word 0 or 4
      uint32_t goff =
          (uint32_t)(((((size_t)ks * (N >> 4) + (bn >> 4) + mc) * 32 + ln) * 8 + w4) * 4);
      uint32_t loff = (uint32_t)(size_t)&Bs[buf][mc * 384 + ln * 12 + w4];
      async_load_b128(loff, Bp, goff);
    }
    // A tile (BM x KT bf16)
    if (MODE == 0) {
      const float* A = (const float*)Aglob;
      int row = t >> 3, kq = (t & 7) * 4;
#pragma unroll
      for (int p = 0; p < 4; ++p) {
        int r = row + p * 32;
        float4 v = *reinterpret_cast<const float4*>(A + (size_t)(bm + r) * K + k0 + kq);
        uint32_t* dst = reinterpret_cast<uint32_t*>(&As[buf][r * LDA + kq]);
        dst[0] = pack_bf16x2(v.x, v.y);
        dst[1] = pack_bf16x2(v.z, v.w);
      }
      if (k0 + 2 * KT < K)  // global_prefetch_b8 two tiles ahead
        __builtin_prefetch(A + (size_t)(bm + (t >> 1)) * K + k0 + 2 * KT + (t & 1) * 16, 0, 1);
    } else {
#pragma unroll
      for (int p = 0; p < 2; ++p) {
        int idx = t + p * 256;
        int r = idx >> 2, kq = (idx & 3) * 8;
        uint32_t goff = (uint32_t)(((size_t)(bm + r) * K + k0 + kq) * 2);
        uint32_t loff = (uint32_t)(size_t)&As[buf][r * LDA + kq];
        async_load_b128(loff, Aglob, goff);
      }
    }
  };

  const int ksteps = K / KT;
  stage(0, 0);
  for (int ks = 0; ks < ksteps; ++ks) {
    const int cur = ks & 1;
    const bool more = (ks + 1 < ksteps);
    if (more) stage(ks + 1, cur ^ 1);  // ping-pong buffer: readers cleared at
                                       // the end-of-iteration barrier of ks-1
    // force only the CURRENT tile's async ops complete; next tile stays in flight
    if (more) { if constexpr (MODE == 0) wait_async_le2(); else wait_async_le4(); }
    else      { wait_async_le0(); }
    __syncthreads();

    // ---- wave-level 32x64 WMMA block ----
    v16bf afrag[2], bfrag[4];
#pragma unroll
    for (int i = 0; i < 2; ++i)
      afrag[i] = load_a_frag(&As[cur][(wm + i * 16) * LDA], LDA, lane);
#pragma unroll
    for (int j = 0; j < 4; ++j) bfrag[j] = load_b_frag(&Bs[cur][0], (wn >> 4) + j, lane);
#pragma unroll
    for (int i = 0; i < 2; ++i)
#pragma unroll
      for (int j = 0; j < 4; ++j)
        acc[i][j] = __builtin_amdgcn_wmma_f32_16x16x32_bf16(
            false, afrag[i], false, bfrag[j], (short)0, acc[i][j], false, false);
    __syncthreads();
  }

  // ---- epilogue ----
  const int mh = (lane >> 4) * 8;  // lanes 16-31 carry M = r+8
  const int nl = lane & 15;
  if (MODE <= 1) {
    // stage bf16 tile in LDS, then coalesced b128 writeback
#pragma unroll
    for (int i = 0; i < 2; ++i)
#pragma unroll
      for (int j = 0; j < 4; ++j) {
        int col = wn + j * 16 + nl;
        float bc = bias[bn + col];
        FragF f; f.v = acc[i][j];
#pragma unroll
        for (int r = 0; r < 8; ++r) {
          float val = f.f[r] + bc;
          if (MODE == 0) val = fmaxf(val, 0.f);
          obuf[(wm + i * 16 + mh + r) * LDO + col] = f32_to_bf16(val);
        }
      }
    __syncthreads();
    uint16_t* Cb = (uint16_t*)Cout;
#pragma unroll
    for (int p = 0; p < 8; ++p) {
      int idx = t + p * 256;               // 128 rows x 16 chunks of 8 bf16
      int row = idx >> 4, c8 = (idx & 15) * 8;
      uint4 v = *reinterpret_cast<const uint4*>(&obuf[row * LDO + c8]);
      *reinterpret_cast<uint4*>(Cb + (size_t)(bm + row) * N + bn + c8) = v;
    }
  } else {
    // stage logits in LDS, per-pair gumbel argmax -> one-hot float2
#pragma unroll
    for (int i = 0; i < 2; ++i)
#pragma unroll
      for (int j = 0; j < 4; ++j) {
        int col = wn + j * 16 + nl;
        float bc = bias[bn + col];
        FragF f; f.v = acc[i][j];
#pragma unroll
        for (int r = 0; r < 8; ++r)
          zbuf[(wm + i * 16 + mh + r) * LDZ + col] = f.f[r] + bc;
      }
    __syncthreads();
    float* Of = (float*)Cout;
#pragma unroll
    for (int p = 0; p < 32; ++p) {
      int idx = t + p * 256;               // 128 rows x 64 pairs
      int row = idx >> 6, pr = idx & 63;
      float l0 = zbuf[row * LDZ + 2 * pr];
      float l1 = zbuf[row * LDZ + 2 * pr + 1];
      size_t gb = (size_t)(bm + row);
      int n0 = bn + 2 * pr;
      float2 uv = *reinterpret_cast<const float2*>(Ug + gb * 1024 + n0);
      float g0 = -__logf(-__logf(uv.x + 1e-20f) + 1e-20f);
      float g1 = -__logf(-__logf(uv.y + 1e-20f) + 1e-20f);
      bool pick0 = (l0 + g0) >= (l1 + g1);  // argmax, ties -> index 0
      float2 o; o.x = pick0 ? 1.f : 0.f; o.y = pick0 ? 0.f : 1.f;
      *reinterpret_cast<float2*>(Of + gb * 1024 + n0) = o;
    }
  }
}

// ---------------------------------------------------------------------------
extern "C" void kernel_launch(void* const* d_in, const int* in_sizes, int n_in,
                              void* d_out, int out_size, void* d_ws, size_t ws_size,
                              hipStream_t stream) {
  const float* s  = (const float*)d_in[0];   // [B, 1024]
  const float* u  = (const float*)d_in[1];   // [B, 512, 2]
  const float* W1 = (const float*)d_in[2];   // [1024, 1024]
  const float* b1 = (const float*)d_in[3];   // [1024]
  const float* W2 = (const float*)d_in[4];   // [1024, 256]
  const float* b2 = (const float*)d_in[5];   // [256]
  const float* hW = (const float*)d_in[6];   // [512, 256, 2]
  const float* hb = (const float*)d_in[7];   // [512, 2] -> flat [1024]

  constexpr int B = 32768, S = 1024, H = 1024, DH = 256, N3 = 1024;

  char* ws = (char*)d_ws;
  uint32_t* W1f = (uint32_t*)ws; ws += (size_t)(S / 2) * H * 4;    // 2 MB
  uint32_t* W2f = (uint32_t*)ws; ws += (size_t)(H / 2) * DH * 4;   // 0.5 MB
  uint32_t* W3f = (uint32_t*)ws; ws += (size_t)(DH / 2) * N3 * 4;  // 0.5 MB
  uint16_t* hbuf  = (uint16_t*)ws; ws += (size_t)B * H * 2;        // 64 MB
  uint16_t* awbuf = (uint16_t*)ws; ws += (size_t)B * DH * 2;       // 16 MB

  pack_w_kernel<<<(S / 2 * H + 255) / 256, 256, 0, stream>>>(W1, W1f, S, H);
  pack_w_kernel<<<(H / 2 * DH + 255) / 256, 256, 0, stream>>>(W2, W2f, H, DH);
  pack_headw_kernel<<<(DH / 2 * N3 + 255) / 256, 256, 0, stream>>>(hW, W3f);

  dim3 g1(B / 128, H / 128);    // 256 x 8
  gemm_wmma<0><<<g1, 256, 0, stream>>>(s, W1f, b1, hbuf, nullptr, S, H);
  dim3 g2(B / 128, DH / 128);   // 256 x 2
  gemm_wmma<1><<<g2, 256, 0, stream>>>(hbuf, W2f, b2, awbuf, nullptr, H, DH);
  dim3 g3(B / 128, N3 / 128);   // 256 x 8
  gemm_wmma<2><<<g3, 256, 0, stream>>>(awbuf, W3f, hb, d_out, u, DH, N3);
}
